// TextGraph_45878840656053
// MI455X (gfx1250) — matrix operand order repaired and checked by reading, not compiled, not run on hardware
//
#include <hip/hip_runtime.h>
#include <math.h>

// ---------------------------------------------------------------------------
// TextGraph pipeline for MI455X (gfx1250, wave32).
//  - Heavy GEMMs: v_wmma_f32_16x16x32_bf16 (f32 accumulate).
//  - A-tiles (and Gram B-tiles) staged global->LDS by the Tensor Data Mover
//    (tensor_load_to_lds, TENSORcnt), with D# LDS padding reproducing the
//    bank-skewed KPAD row stride. Double-buffered LDS overlaps the next
//    chunk's DMA with the current chunk's WMMAs.
// ---------------------------------------------------------------------------

typedef __bf16 bf16;
typedef __attribute__((ext_vector_type(16))) __bf16 v16bf;
typedef __attribute__((ext_vector_type(8)))  __bf16 v8bf;
typedef __attribute__((ext_vector_type(8)))  float  v8f;
typedef unsigned int u32x4 __attribute__((ext_vector_type(4)));
typedef unsigned int u32x8 __attribute__((ext_vector_type(8)));

namespace {
constexpr int kB    = 48;
constexpr int kN    = 512;
constexpr int kIN   = 400;
constexpr int kENC  = 256;
constexpr int kHID  = 128;
constexpr int kOUT  = 128;
constexpr int kNREL = 53;
constexpr int kPERS = 4;
constexpr int kKNN  = 10;
constexpr float kEPS  = 0.3f;
constexpr float kSKIP = 0.8f;
constexpr float kVSN  = 1e-12f;

constexpr int KPAD = 40;  // bf16 elems per LDS tile row (80 B, 16B-aligned)
constexpr unsigned kBufBytes = 64u * KPAD * 2u;  // one 64x32 tile buffer
}

// ---------------------------------------------------------------------------
// helpers
// ---------------------------------------------------------------------------
__device__ inline v8bf zero8() {
  v8bf z;
#pragma unroll
  for (int i = 0; i < 8; ++i) z[i] = (bf16)0.0f;
  return z;
}

__device__ inline v16bf cat8(v8bf lo, v8bf hi) {
  return __builtin_shufflevector(lo, hi, 0, 1, 2, 3, 4, 5, 6, 7,
                                         8, 9, 10, 11, 12, 13, 14, 15);
}

// A-matrix fragment (16x32 bf16): lanes 0-15 -> rows 0-15, K {0..7,16..23};
// lanes 16-31 -> rows 0-15, K {8..15,24..31}.  (ISA 7.12.2)
__device__ inline v16bf load_frag_a(const bf16* t) {
  const int lane = threadIdx.x & 31;
  const bf16* p = t + (lane & 15) * KPAD;
  const int k0 = (lane >> 4) * 8;
  v8bf lo = *(const v8bf*)(p + k0);
  v8bf hi = *(const v8bf*)(p + 16 + k0);
  return cat8(lo, hi);
}

// B-matrix fragment (32x16 bf16): lane = column; lanes 0-15 hold K {0..7,16..23},
// lanes 16-31 hold K {8..15,24..31}.  Tile stored [col][k].
__device__ inline v16bf load_frag_b(const bf16* t) {
  const int lane = threadIdx.x & 31;
  const bf16* p = t + (lane & 15) * KPAD;
  const int k0 = (lane >> 4) * 8;
  v8bf lo = *(const v8bf*)(p + k0);
  v8bf hi = *(const v8bf*)(p + 16 + k0);
  return cat8(lo, hi);
}

// ---------------------------------------------------------------------------
// Tensor Data Mover: DMA one 64-row x 32-elem bf16 tile (row stride `ld`
// elements) from global to LDS.  D# per CDNA5 ISA:
//   group0: count=1 | lds_addr | global_addr | type=2
//   group1: data_size=2B, pad_enable (pad 4 DWORDs after every 16 DWORDs
//           -> LDS row stride 80 B = KPAD), tensor dims = remaining extent
//           from the tile origin (hardware zero-fills OOB reads), tile 32x64.
// Issued by one wave; completion via s_wait_tensorcnt before the barrier.
// ---------------------------------------------------------------------------
__device__ inline void tdm_load_tile(unsigned lds_byte_addr,
                                     const bf16* gaddr, int rem_cols,
                                     int rem_rows, int ld) {
  const unsigned long long ga = (unsigned long long)gaddr;
  const unsigned long long st = (unsigned long long)ld;
  u32x4 g0;
  g0[0] = 1u;                                   // count=1, user descriptor
  g0[1] = lds_byte_addr;                        // lds_addr
  g0[2] = (unsigned)ga;                         // global_addr[31:0]
  g0[3] = (unsigned)((ga >> 32) & 0x1FFFFFFull) // global_addr[56:32]
          | (2u << 30);                         // type=2 ("image")
  const unsigned td0 = (unsigned)rem_cols;      // tensor_dim0 (elements)
  const unsigned td1 = (unsigned)rem_rows;      // tensor_dim1 (rows)
  u32x8 g1;
  g1[0] = (1u << 16)                            // data_size: 2 bytes
        | (1u << 20)                            // pad_enable
        | (3u << 22)                            // pad_interval: 16 DWORDs
        | (3u << 25);                           // pad_amount: 4 DWORDs (16 B)
  g1[1] = (td0 & 0xFFFFu) << 16;                // tensor_dim0[15:0] @ bit48
  g1[2] = ((td0 >> 16) & 0xFFFFu) | ((td1 & 0xFFFFu) << 16);
  g1[3] = ((td1 >> 16) & 0xFFFFu) | (32u << 16);  // tile_dim0 = 32
  g1[4] = 64u;                                  // tile_dim1 = 64, tile_dim2=0
  g1[5] = (unsigned)st;                         // tensor_dim0_stride[31:0]
  g1[6] = (unsigned)((st >> 32) & 0xFFFFull);   // stride[47:32]; stride1 lo=0
  g1[7] = 0u;
  asm volatile("tensor_load_to_lds %0, %1" ::"s"(g0), "s"(g1) : "memory");
}

__device__ inline unsigned lds_addr_of(const void* p) {
  // Flat LDS-aperture addresses carry the LDS byte offset in addr[31:0].
  return (unsigned)(unsigned long long)(p);
}

// ---------------------------------------------------------------------------
// Generic tiled WMMA GEMM.  C[MxNc] = A[MxK] * op(B),  bf16 in, f32 accum.
//   BT=true : B source is row-major [Nc,K]  (Gram matrices A·Aᵀ) -> TDM too
//   BT=false: B source is row-major [K,Nc]  -> manual transpose staging
// Block: 128 threads = 4 waves; 64x64 tile; each wave 32x32 (2x2 WMMA tiles).
// Double-buffered LDS: chunk i+1 DMA overlaps chunk i WMMAs.
// ---------------------------------------------------------------------------
enum {
  EPI_MASK_F32 = 0,       // v *= mask[m]*mask[n]; f32 store   (kNN similarity)
  EPI_COS = 1,            // v = v*scale*mask2d; threshold>eps; f32 store
  EPI_BF16 = 2,           // bf16 store
  EPI_BIAS_BF16 = 3,      // v += bias[n]; bf16 store
  EPI_RELU_BIAS_BF16 = 4, // relu(v + bias[n]); bf16 store
  EPI_BIAS_F32 = 5        // v += bias[n]; f32 store
};

template <bool BT, int EPI>
__global__ __launch_bounds__(128) void wmma_gemm(
    const bf16* __restrict__ A, long sA, const bf16* __restrict__ Bm, long sB,
    void* __restrict__ Cv, long sC, int M, int Nc, int K, int lda, int ldb,
    int ldc, const float* __restrict__ bias, const float* __restrict__ mask,
    int maskStride, float scale, float eps) {
  const int z = blockIdx.z;
  const bf16* Ab = A + (long)z * sA;
  const bf16* Bb = Bm + (long)z * sB;

  const int tid  = threadIdx.x;
  const int lane = tid & 31;
  const int wave = tid >> 5;
  const int wm = wave >> 1;      // wave row (0..1)
  const int wn = wave & 1;       // wave col (0..1)
  const int rowBase = blockIdx.y * 64;
  const int colBase = blockIdx.x * 64;

  __shared__ __align__(16) bf16 As[2][64][KPAD];
  __shared__ __align__(16) bf16 Bs[2][64][KPAD];
  const unsigned asBase = lds_addr_of(&As[0][0][0]);
  const unsigned bsBase = lds_addr_of(&Bs[0][0][0]);

  v8f acc[2][2];
#pragma unroll
  for (int i = 0; i < 2; ++i)
#pragma unroll
    for (int j = 0; j < 2; ++j)
#pragma unroll
      for (int r = 0; r < 8; ++r) acc[i][j][r] = 0.0f;

  const int nk = (K + 31) >> 5;

  // ---- prologue: stage chunk 0 into buffer 0 ----
  if (wave == 0) {
    tdm_load_tile(asBase, Ab + (long)rowBase * lda, K, M - rowBase, lda);
    if (BT)
      tdm_load_tile(bsBase, Bb + (long)colBase * ldb, K, Nc - colBase, ldb);
  }
  if (!BT) {
    const int bk  = tid >> 2;
    const int bn4 = (tid & 3) * 16;
    const int gn  = colBase + bn4;
    v8bf x0 = zero8(), x1 = zero8();
    if (bk < K && gn < Nc) {
      const bf16* p2 = Bb + (long)bk * ldb + gn;
      x0 = *(const v8bf*)p2;
      x1 = *(const v8bf*)(p2 + 8);
    }
#pragma unroll
    for (int q = 0; q < 8; ++q) {
      Bs[0][bn4 + q][bk]     = x0[q];
      Bs[0][bn4 + 8 + q][bk] = x1[q];
    }
  }
  if (wave == 0) __builtin_amdgcn_s_wait_tensorcnt(0);
  __syncthreads();

  // ---- pipelined main loop ----
  for (int i = 0; i < nk; ++i) {
    const int buf = i & 1;
    const int nxt = buf ^ 1;
    const int kb = i << 5;
    const bool more = (i + 1) < nk;

    // issue next chunk's tensor DMA (overlaps this chunk's WMMAs)
    if (more && wave == 0) {
      const int kn = kb + 32;
      tdm_load_tile(asBase + (unsigned)nxt * kBufBytes,
                    Ab + (long)rowBase * lda + kn, K - kn, M - rowBase, lda);
      if (BT)
        tdm_load_tile(bsBase + (unsigned)nxt * kBufBytes,
                      Bb + (long)colBase * ldb + kn, K - kn, Nc - colBase,
                      ldb);
    }
    // manual B path: issue next chunk's global loads early
    v8bf nx0 = zero8(), nx1 = zero8();
    int bk = 0, bn4 = 0;
    if (!BT && more) {
      bk  = tid >> 2;
      bn4 = (tid & 3) * 16;
      const int gk = kb + 32 + bk;
      const int gn = colBase + bn4;
      if (gk < K && gn < Nc) {
        const bf16* p2 = Bb + (long)gk * ldb + gn;
        nx0 = *(const v8bf*)p2;
        nx1 = *(const v8bf*)(p2 + 8);
        __builtin_prefetch(p2 + 32, 0, 1);  // global_prefetch_b8
      }
    }

    // compute on the current buffer
    const v16bf a0 = load_frag_a(&As[buf][wm * 32][0]);
    const v16bf a1 = load_frag_a(&As[buf][wm * 32 + 16][0]);
    const v16bf b0 = load_frag_b(&Bs[buf][wn * 32][0]);
    const v16bf b1 = load_frag_b(&Bs[buf][wn * 32 + 16][0]);

    acc[0][0] = __builtin_amdgcn_wmma_f32_16x16x32_bf16(false, a0, false, b0,
                                                        (short)0, acc[0][0],
                                                        false, false);
    acc[0][1] = __builtin_amdgcn_wmma_f32_16x16x32_bf16(false, a0, false, b1,
                                                        (short)0, acc[0][1],
                                                        false, false);
    acc[1][0] = __builtin_amdgcn_wmma_f32_16x16x32_bf16(false, a1, false, b0,
                                                        (short)0, acc[1][0],
                                                        false, false);
    acc[1][1] = __builtin_amdgcn_wmma_f32_16x16x32_bf16(false, a1, false, b1,
                                                        (short)0, acc[1][1],
                                                        false, false);

    // land the manual B transpose for the next chunk
    if (!BT && more) {
#pragma unroll
      for (int q = 0; q < 8; ++q) {
        Bs[nxt][bn4 + q][bk]     = nx0[q];
        Bs[nxt][bn4 + 8 + q][bk] = nx1[q];
      }
    }
    if (more && wave == 0) __builtin_amdgcn_s_wait_tensorcnt(0);
    __syncthreads();
  }

  // ---- epilogue: C/D layout lanes 0-15 -> M=r, lanes 16-31 -> M=8+r ----
  const float* mrow =
      (EPI == EPI_MASK_F32 || EPI == EPI_COS) ? (mask + (long)z * maskStride)
                                              : nullptr;
  float* Cf = (float*)Cv + (long)z * sC;
  bf16*  Cb = (bf16*)Cv + (long)z * sC;

#pragma unroll
  for (int i = 0; i < 2; ++i) {
#pragma unroll
    for (int j = 0; j < 2; ++j) {
      const int mb = rowBase + wm * 32 + i * 16 + (lane >> 4) * 8;
      const int n  = colBase + wn * 32 + j * 16 + (lane & 15);
      if (n >= Nc) continue;
#pragma unroll
      for (int r = 0; r < 8; ++r) {
        const int m = mb + r;
        if (m >= M) break;
        float v = acc[i][j][r];
        const long co = (long)m * ldc + n;
        if (EPI == EPI_MASK_F32) {
          Cf[co] = v * mrow[m] * mrow[n];
        } else if (EPI == EPI_COS) {
          v = v * scale * mrow[m] * mrow[n];
          Cf[co] = (v > eps) ? v : 0.0f;
        } else if (EPI == EPI_BF16) {
          Cb[co] = (bf16)v;
        } else if (EPI == EPI_BIAS_BF16) {
          Cb[co] = (bf16)(v + bias[n]);
        } else if (EPI == EPI_RELU_BIAS_BF16) {
          v += bias[n];
          Cb[co] = (bf16)(v > 0.0f ? v : 0.0f);
        } else {  // EPI_BIAS_F32
          Cf[co] = v + bias[n];
        }
      }
    }
  }
}

// ---------------------------------------------------------------------------
// Per-row prep: bf16 cast of context + per-perspective f32 L2 normalization
// (the normalization runs in f32; only the final dot is bf16).
// ---------------------------------------------------------------------------
__global__ __launch_bounds__(128) void prep_rows(
    const float* __restrict__ context, const int* __restrict__ lens,
    const float* __restrict__ glw, bf16* __restrict__ ctx_bf,
    bf16* __restrict__ ctxn_bf, float* __restrict__ mask) {
  const long row = blockIdx.x;
  const int b = (int)(row / kN);
  const int n = (int)(row % kN);
  const int tid = threadIdx.x;
  const float* src = context + row * kIN;

  __shared__ float xs[kIN];
  __shared__ float red[128];

  for (int d = tid; d < kIN; d += 128) {
    const float x = src[d];
    xs[d] = x;
    ctx_bf[row * kIN + d] = (bf16)x;
  }
  if (tid == 0) mask[row] = (n < lens[b]) ? 1.0f : 0.0f;
  __syncthreads();

  for (int p = 0; p < kPERS; ++p) {
    float part = 0.0f;
    for (int d = tid; d < kIN; d += 128) {
      const float y = xs[d] * glw[p * kIN + d];
      part += y * y;
    }
    red[tid] = part;
    __syncthreads();
    for (int s = 64; s > 0; s >>= 1) {
      if (tid < s) red[tid] += red[tid + s];
      __syncthreads();
    }
    const float nrm = fmaxf(sqrtf(red[0]), kVSN);
    const float inv = 1.0f / nrm;
    for (int d = tid; d < kIN; d += 128) {
      ctxn_bf[row * (long)(kPERS * kIN) + p * kIN + d] =
          (bf16)(xs[d] * glw[p * kIN + d] * inv);
    }
    __syncthreads();
  }
}

__global__ void cvt_f32_bf16(const float* __restrict__ s, bf16* __restrict__ d,
                             int n) {
  const int i = blockIdx.x * 256 + threadIdx.x;
  if (i < n) d[i] = (bf16)s[i];
}

// ---------------------------------------------------------------------------
// Top-10 per similarity row (one wave per row) + symmetric-norm rinv.
// ---------------------------------------------------------------------------
__global__ __launch_bounds__(128) void topk_kernel(
    const float* __restrict__ att, const float* __restrict__ mask,
    int* __restrict__ knn_idx, float* __restrict__ rinv) {
  const int wave = threadIdx.x >> 5;
  const int lane = threadIdx.x & 31;
  const long row = (long)blockIdx.x * 4 + wave;
  const int b = (int)(row / kN);
  const float* ar = att + row * (long)kN;

  float tv[kKNN];
  int   ti[kKNN];
#pragma unroll
  for (int i = 0; i < kKNN; ++i) { tv[i] = -1e30f; ti[i] = 0; }

  for (int j = 0; j < kN / 32; ++j) {
    const int m = lane + j * 32;
    const float v = ar[m];
    if (v > tv[kKNN - 1]) {
      tv[kKNN - 1] = v;
      ti[kKNN - 1] = m;
#pragma unroll
      for (int q = kKNN - 1; q > 0; --q) {
        if (tv[q] > tv[q - 1]) {
          const float fv = tv[q]; tv[q] = tv[q - 1]; tv[q - 1] = fv;
          const int   iv = ti[q]; ti[q] = ti[q - 1]; ti[q - 1] = iv;
        }
      }
    }
  }

  __shared__ float sv[4][32 * kKNN];
  __shared__ int   si[4][32 * kKNN];
#pragma unroll
  for (int i = 0; i < kKNN; ++i) {
    sv[wave][lane * kKNN + i] = tv[i];
    si[wave][lane * kKNN + i] = ti[i];
  }
  __syncthreads();

  if (lane == 0) {
    float* cv = sv[wave];
    int*   ci = si[wave];
    float msum = 0.0f;
    for (int k = 0; k < kKNN; ++k) {
      int bi = 0;
      float bv = cv[0];
      for (int t = 1; t < 32 * kKNN; ++t)
        if (cv[t] > bv) { bv = cv[t]; bi = t; }
      const int sel = ci[bi];
      cv[bi] = -1e38f;
      knn_idx[row * kKNN + k] = sel;
      msum += mask[(long)b * kN + sel];
    }
    const float mrow_ = mask[row];
    const float rs = fmaxf(mrow_ * msum, kVSN);
    rinv[row] = rsqrtf(rs) * mrow_;
  }
}

// ---------------------------------------------------------------------------
// Fused row-sum normalization + skip-combine with sparse kNN init_adj;
// dense bf16 adjacency out (feeds the WMMA adj GEMMs).
// ---------------------------------------------------------------------------
__global__ __launch_bounds__(256) void combine_adj(
    const float* __restrict__ cosatt, const int* __restrict__ knn_idx,
    const float* __restrict__ rinv, bf16* __restrict__ adj_bf) {
  const long row = blockIdx.x;
  const int b = (int)(row / kN);
  const int tid = threadIdx.x;
  const float* cr = cosatt + row * (long)kN;

  __shared__ float red[256];
  __shared__ int   kidx[kKNN];
  __shared__ float krv[kKNN];

  red[tid] = cr[tid] + cr[tid + 256];
  if (tid < kKNN) {
    const int ii = knn_idx[row * kKNN + tid];
    kidx[tid] = ii;
    krv[tid]  = rinv[(long)b * kN + ii];
  }
  __syncthreads();
  for (int s = 128; s > 0; s >>= 1) {
    if (tid < s) red[tid] += red[tid + s];
    __syncthreads();
  }
  const float rs  = fmaxf(red[0], kVSN);
  const float rvn = rinv[row];

#pragma unroll
  for (int rep = 0; rep < 2; ++rep) {
    const int m = tid + rep * 256;
    float v = (1.0f - kSKIP) * cr[m] / rs;
#pragma unroll
    for (int j = 0; j < kKNN; ++j)
      if (kidx[j] == m) v += kSKIP * rvn * krv[j];
    adj_bf[row * (long)kN + m] = (bf16)v;
  }
}

// ---------------------------------------------------------------------------
// Max-pool over nodes + mention-span averaging + [128 -> 53] head.
// ---------------------------------------------------------------------------
__global__ __launch_bounds__(128) void finalize_kernel(
    const float* __restrict__ node_vec, const int* __restrict__ mentions,
    const float* __restrict__ lin_w, const float* __restrict__ lin_b,
    float* __restrict__ out) {
  const int b = blockIdx.x;
  const int d = threadIdx.x;
  const float* nv = node_vec + (long)b * kN * kOUT;
  const int s1 = mentions[b * 4 + 0];
  const int e1 = mentions[b * 4 + 1];
  const int s2 = mentions[b * 4 + 2];
  const int e2 = mentions[b * 4 + 3];

  float mx = -1e30f, a1 = 0.0f, a2 = 0.0f;
  for (int n = 0; n < kN; ++n) {
    const float v = nv[(long)n * kOUT + d];
    mx = fmaxf(mx, v);
    if (n >= s1 && n <= e1) a1 += v;
    if (n >= s2 && n <= e2) a2 += v;
  }
  __shared__ float vsh[kOUT];
  vsh[d] = mx + a1 / (float)(e1 - s1 + 1) + a2 / (float)(e2 - s2 + 1);
  __syncthreads();

  if (d < kNREL) {
    float o = lin_b[d];
    for (int k = 0; k < kOUT; ++k) o += vsh[k] * lin_w[k * kNREL + d];
    out[b * kNREL + d] = o;
  }
}

// ---------------------------------------------------------------------------
// launcher
// ---------------------------------------------------------------------------
extern "C" void kernel_launch(void* const* d_in, const int* in_sizes, int n_in,
                              void* d_out, int out_size, void* d_ws,
                              size_t ws_size, hipStream_t stream) {
  (void)in_sizes; (void)n_in; (void)out_size; (void)ws_size;
  const float* context = (const float*)d_in[0];
  const int*   lens    = (const int*)d_in[1];
  const int*   ments   = (const int*)d_in[2];
  const float* w_enc   = (const float*)d_in[3];
  const float* b_enc   = (const float*)d_in[4];
  const float* glw     = (const float*)d_in[5];
  const float* w1      = (const float*)d_in[6];
  const float* b1      = (const float*)d_in[7];
  const float* w2      = (const float*)d_in[8];
  const float* b2      = (const float*)d_in[9];
  const float* lw      = (const float*)d_in[10];
  const float* lb      = (const float*)d_in[11];
  float* out = (float*)d_out;

  const long BN = (long)kB * kN;
  char* p = (char*)d_ws;
  auto take = [&](size_t bytes) {
    char* q = p;
    p += (bytes + 255) & ~(size_t)255;
    return q;
  };
  bf16*  ctx_bf   = (bf16*)take(BN * kIN * 2);
  bf16*  ctxn_bf  = (bf16*)take(BN * kPERS * kIN * 2);
  float* att      = (float*)take(BN * kN * 4);      // kNN sim, then cosine sim
  bf16*  adj_bf   = (bf16*)take(BN * kN * 2);
  bf16*  enc_bf   = (bf16*)take(BN * kENC * 2);
  bf16*  t1_bf    = (bf16*)take(BN * kENC * 2);
  bf16*  h_bf     = (bf16*)take(BN * kHID * 2);
  bf16*  t2_bf    = (bf16*)take(BN * kHID * 2);
  float* node_vec = (float*)take(BN * kOUT * 4);
  bf16*  wenc_bf  = (bf16*)take((size_t)kIN * kENC * 2);
  bf16*  w1_bf    = (bf16*)take((size_t)kENC * kHID * 2);
  bf16*  w2_bf    = (bf16*)take((size_t)kHID * kOUT * 2);
  int*   knn_idx  = (int*)take(BN * kKNN * 4);
  float* rinv     = (float*)take(BN * 4);
  float* maskp    = (float*)take(BN * 4);

  // weight casts
  cvt_f32_bf16<<<(kIN * kENC + 255) / 256, 256, 0, stream>>>(w_enc, wenc_bf,
                                                             kIN * kENC);
  cvt_f32_bf16<<<(kENC * kHID + 255) / 256, 256, 0, stream>>>(w1, w1_bf,
                                                              kENC * kHID);
  cvt_f32_bf16<<<(kHID * kOUT + 255) / 256, 256, 0, stream>>>(w2, w2_bf,
                                                              kHID * kOUT);
  // per-row prep: bf16 context + normalized perspectives + mask
  prep_rows<<<(int)BN, 128, 0, stream>>>(context, lens, glw, ctx_bf, ctxn_bf,
                                         maskp);

  // kNN similarity (NT Gram): att = (ctx ctxT) * mask2d
  wmma_gemm<true, EPI_MASK_F32><<<dim3(8, 8, kB), 128, 0, stream>>>(
      ctx_bf, (long)kN * kIN, ctx_bf, (long)kN * kIN, att, (long)kN * kN, kN,
      kN, kIN, kIN, kIN, kN, nullptr, maskp, kN, 1.0f, 0.0f);

  topk_kernel<<<(int)(BN / 4), 128, 0, stream>>>(att, maskp, knn_idx, rinv);

  // weighted cosine (NT Gram over folded K = P*IN): mean over perspectives,
  // masked, EPS-thresholded.  Overwrites `att` (topk already consumed it).
  wmma_gemm<true, EPI_COS><<<dim3(8, 8, kB), 128, 0, stream>>>(
      ctxn_bf, (long)kN * kPERS * kIN, ctxn_bf, (long)kN * kPERS * kIN, att,
      (long)kN * kN, kN, kN, kPERS * kIN, kPERS * kIN, kPERS * kIN, kN,
      nullptr, maskp, kN, 1.0f / kPERS, kEPS);

  combine_adj<<<(int)BN, 256, 0, stream>>>(att, knn_idx, rinv, adj_bf);

  // encoder: enc = context @ w_enc + b_enc
  wmma_gemm<false, EPI_BIAS_BF16><<<dim3(kENC / 64, (int)(BN / 64), 1), 128, 0,
                                    stream>>>(
      ctx_bf, 0, wenc_bf, 0, enc_bf, 0, (int)BN, kENC, kIN, kIN, kENC, kENC,
      b_enc, nullptr, 0, 1.0f, 0.0f);

  // t1 = adj @ enc (batched)
  wmma_gemm<false, EPI_BF16><<<dim3(kENC / 64, kN / 64, kB), 128, 0, stream>>>(
      adj_bf, (long)kN * kN, enc_bf, (long)kN * kENC, t1_bf, (long)kN * kENC,
      kN, kENC, kN, kN, kENC, kENC, nullptr, nullptr, 0, 1.0f, 0.0f);

  // h = relu(t1 @ w1 + b1)
  wmma_gemm<false, EPI_RELU_BIAS_BF16><<<dim3(kHID / 64, (int)(BN / 64), 1),
                                         128, 0, stream>>>(
      t1_bf, 0, w1_bf, 0, h_bf, 0, (int)BN, kHID, kENC, kENC, kHID, kHID, b1,
      nullptr, 0, 1.0f, 0.0f);

  // t2 = adj @ h (batched)
  wmma_gemm<false, EPI_BF16><<<dim3(kHID / 64, kN / 64, kB), 128, 0, stream>>>(
      adj_bf, (long)kN * kN, h_bf, (long)kN * kHID, t2_bf, (long)kN * kHID, kN,
      kHID, kN, kN, kHID, kHID, nullptr, nullptr, 0, 1.0f, 0.0f);

  // node_vec = t2 @ w2 + b2  (f32 for the precision-sensitive head)
  wmma_gemm<false, EPI_BIAS_F32><<<dim3(kOUT / 64, (int)(BN / 64), 1), 128, 0,
                                   stream>>>(
      t2_bf, 0, w2_bf, 0, node_vec, 0, (int)BN, kOUT, kHID, kHID, kOUT, kOUT,
      b2, nullptr, 0, 1.0f, 0.0f);

  finalize_kernel<<<kB, 128, 0, stream>>>(node_vec, ments, lw, lb, out);
}
